// MultiLayerParamNonHarmonicEncoding_13460427506363
// MI455X (gfx1250) — compile-verified, matrix operand order checked by reading, not run on hardware
//
#include <hip/hip_runtime.h>
#include <hip/hip_bf16.h>

typedef __attribute__((ext_vector_type(16))) _Float16 v16h;
typedef __attribute__((ext_vector_type(8)))  float    v8f;

#define NTOT_K 4
#define NTOT_L 2
#define NTOT_F 64
#define FANIN  (1 + 2 * NTOT_F)   // 129
#define INV_2PI 0.15915494309189535f

#if __has_builtin(__builtin_amdgcn_sinf) && __has_builtin(__builtin_amdgcn_cosf)
// raw v_sin_f32 / v_cos_f32: input in revolutions (radians * 1/2pi)
__device__ __forceinline__ float hw_sin_rev(float rev) { return __builtin_amdgcn_sinf(rev); }
__device__ __forceinline__ float hw_cos_rev(float rev) { return __builtin_amdgcn_cosf(rev); }
#else
__device__ __forceinline__ float hw_sin_rev(float rev) { return __sinf(rev * 6.283185307179586f); }
__device__ __forceinline__ float hw_cos_rev(float rev) { return __cosf(rev * 6.283185307179586f); }
#endif

// ---------------------------------------------------------------------------
// Setup kernel (1 block, 128 threads): tiny O(16K) work.
//  - freqs[l,f] = 50 * tanh((cumsum_f(softplus(fd)+0.25) + bias[l]) / 50),
//    stored PRE-SCALED by 1/(2pi) so the main kernel can feed raw v_sin_f32.
//  - bias_out[k] = sum_l (W[l,k,0] + b[l,k])
//  - pack block-diagonal B matrix (1024 x 16, f16) into the exact CDNA5
//    16-bit B-operand register layout, 32 K-chunks of 32, so the main kernel
//    can load each lane's v16h with one 32B LDS read.
//    Chunk index c = (((l*2 + fbi)*4 + kk)*2 + tt), fbi selects f-half of 32.
// ---------------------------------------------------------------------------
__global__ void mlpnhe_setup(const float* __restrict__ fd,
                             const float* __restrict__ fbias,
                             const float* __restrict__ W,
                             const float* __restrict__ b,
                             float* __restrict__ ws_freqs,   // 128 floats (rev units)
                             float* __restrict__ ws_bias,    // 4 floats
                             _Float16* __restrict__ ws_b)    // 32*32*16 halves
{
    __shared__ float s_delta[NTOT_L * NTOT_F];
    __shared__ float s_freq[NTOT_L * NTOT_F];
    const int t = threadIdx.x;   // 128 threads

    if (t < NTOT_L * NTOT_F) {
        float x = fd[t];
        s_delta[t] = logf(1.0f + expf(x)) + 0.25f;   // softplus + DELTA_MIN
    }
    __syncthreads();
    if (t == 0 || t == NTOT_F) {            // one serial cumsum per layer
        const float fb = fbias[t / NTOT_F];
        float acc = 0.0f;
        for (int f = 0; f < NTOT_F; ++f) {
            acc += s_delta[t + f];
            // radians -> revolutions fold: * 1/(2pi)
            s_freq[t + f] = 50.0f * tanhf((acc + fb) * (1.0f / 50.0f)) * INV_2PI;
        }
    }
    __syncthreads();
    if (t < NTOT_L * NTOT_F) ws_freqs[t] = s_freq[t];
    if (t < NTOT_K) {
        ws_bias[t] = W[0 * NTOT_K * FANIN + t * FANIN] + b[t]
                   + W[1 * NTOT_K * FANIN + t * FANIN] + b[NTOT_K + t];
    }

    // Pack B: flat half index = ((c*32 + lane)*16 + slot)
    for (int idx = t; idx < 32 * 32 * 16; idx += 128) {
        const int c    = idx >> 9;
        const int rem  = idx & 511;
        const int lane = rem >> 4;
        const int j    = rem & 15;          // half slot within v16h
        const int l    = c >> 4;
        const int fbi  = (c >> 3) & 1;
        const int kk   = (c >> 1) & 3;
        const int tt   = c & 1;             // 0=sin, 1=cos
        const int col  = lane & 15;         // B column (N)
        // CDNA5 16-bit operand layout: slot j -> K within the 32-chunk
        const int v    = j >> 1, h = j & 1;
        const int base = (v < 4) ? ((lane < 16) ? 0 : 8)
                                 : ((lane < 16) ? 16 : 24);
        const int K    = base + 2 * (v & 3) + h;
        const int f    = fbi * 32 + K;
        float w = 0.0f;
        if (col == kk)
            w = W[l * NTOT_K * FANIN + kk * FANIN + 1 + tt * NTOT_F + f];
        ws_b[idx] = (_Float16)w;
    }
}

// ---------------------------------------------------------------------------
// Main kernel: one 16-row n-tile per wave iteration.
// Per tile: 512 v_sin/v_cos per lane + 32 v_wmma_f32_16x16x32_f16.
// ---------------------------------------------------------------------------
__global__ void __launch_bounds__(256)
mlpnhe_main(const float* __restrict__ e,
            float* __restrict__ out,
            const float* __restrict__ ws_freqs,
            const float* __restrict__ ws_bias,
            const _Float16* __restrict__ ws_b,
            int Ntot, int numTiles)
{
    __shared__ v16h  lds_b[32 * 32];           // 32 KB packed B chunks
    __shared__ float lds_freqs[NTOT_L * NTOT_F];
    __shared__ float lds_bias[NTOT_K];

    const int tid = threadIdx.x;
    {   // cooperative LDS fill (uint4 = 16B granules)
        const uint4* src = (const uint4*)ws_b;
        uint4*       dst = (uint4*)lds_b;
        for (int i = tid; i < (32 * 32 * 16 * 2) / 16; i += 256) dst[i] = src[i];
        if (tid < NTOT_L * NTOT_F) lds_freqs[tid] = ws_freqs[tid];
        if (tid < NTOT_K)          lds_bias[tid]  = ws_bias[tid];
    }
    __syncthreads();

    const int lane = tid & 31;
    const int wave = tid >> 5;
    const int waveGlobal = blockIdx.x * 8 + wave;
    const int waveStride = gridDim.x * 8;
    const int m  = lane & 15;                   // A-matrix row this lane feeds
    const int lo = (lane < 16) ? 0 : 8;         // K sub-ranges per wave32 layout
    const int hi = (lane < 16) ? 16 : 24;

    for (int tile = waveGlobal; tile < numTiles; tile += waveStride) {
        const int n0   = tile * 16;
        const int nIdx = n0 + m;
        const float ev = e[(nIdx < Ntot) ? nIdx : (Ntot - 1)];

        float p[NTOT_K];                        // e^1 .. e^4
        p[0] = ev; p[1] = ev * ev; p[2] = p[1] * ev; p[3] = p[2] * ev;

        v8f acc = {};
#pragma unroll
        for (int l = 0; l < NTOT_L; ++l) {
#pragma unroll
            for (int fbi = 0; fbi < 2; ++fbi) {
                float fr[16];                   // pre-scaled freqs (revolutions)
#pragma unroll
                for (int i = 0; i < 8; ++i) {
                    fr[i]     = lds_freqs[l * NTOT_F + fbi * 32 + lo + i];
                    fr[8 + i] = lds_freqs[l * NTOT_F + fbi * 32 + hi + i];
                }
#pragma unroll
                for (int kk = 0; kk < NTOT_K; ++kk) {
                    const float pw = p[kk];
                    float ph[16];               // phase in revolutions
#pragma unroll
                    for (int i = 0; i < 16; ++i) ph[i] = pw * fr[i];
#pragma unroll
                    for (int tt = 0; tt < 2; ++tt) {
                        v16h a;
#pragma unroll
                        for (int i = 0; i < 16; ++i)
                            a[i] = (_Float16)(tt ? hw_cos_rev(ph[i])
                                                 : hw_sin_rev(ph[i]));
                        const int c = (((l * 2 + fbi) * 4 + kk) * 2 + tt);
                        const v16h bb = lds_b[c * 32 + lane];
                        acc = __builtin_amdgcn_wmma_f32_16x16x32_f16(
                                  false, a, false, bb, (short)0, acc,
                                  false, false);
                    }
                }
            }
        }

        // C layout: VGPR i / lanes0-15 -> (M=i, N=lane); lanes16-31 -> (M=i+8)
        const int col = lane & 15;
        if (col < NTOT_K) {
            const float bias  = lds_bias[col];
            const int   rbase = n0 + ((lane < 16) ? 0 : 8);
#pragma unroll
            for (int i = 0; i < 8; ++i) {
                const int n = rbase + i;
                if (n < Ntot)
                    out[col * Ntot + n] = acc[i] * 0.125f + bias;  // / sqrt(64)
            }
        }
    }
}

// ---------------------------------------------------------------------------
extern "C" void kernel_launch(void* const* d_in, const int* in_sizes, int n_in,
                              void* d_out, int out_size, void* d_ws, size_t ws_size,
                              hipStream_t stream) {
    const float* e     = (const float*)d_in[0];
    const float* fd    = (const float*)d_in[1];
    const float* fbias = (const float*)d_in[2];
    const float* W     = (const float*)d_in[3];
    const float* b     = (const float*)d_in[4];
    float*       out   = (float*)d_out;

    const int Ntot     = in_sizes[0];
    const int numTiles = (Ntot + 15) / 16;

    // workspace layout: 128 freqs | 4 bias | 4 pad | packed B (16B aligned)
    float*    ws_freqs = (float*)d_ws;
    float*    ws_bias  = ws_freqs + 128;
    _Float16* ws_b     = (_Float16*)(ws_freqs + 136);   // byte offset 544 (16B aligned)

    mlpnhe_setup<<<1, 128, 0, stream>>>(fd, fbias, W, b, ws_freqs, ws_bias, ws_b);

    const int blocks = 1024;   // 8192 waves, grid-stride over 31250 tiles
    mlpnhe_main<<<blocks, 256, 0, stream>>>(e, out, ws_freqs, ws_bias, ws_b,
                                            Ntot, numTiles);
}
